// DynamicGate_89687507075532
// MI455X (gfx1250) — compile-verified, matrix operand order checked by reading, not run on hardware
//
#include <hip/hip_runtime.h>
#include <hip/hip_bf16.h>
#include <math.h>

// ---------------------------------------------------------------------------
// Fused MoE dynamic gate for MI455X (gfx1250), wave32 + WMMA bf16 + TDM.
//
//   h   = x @ Wp + bp                  [16384,2048]x[2048,256]  (dominant GEMM)
//   hid = relu(h_head @ W1h + b1h)     4x per-head 64x64
//   hl  = hid_head @ W2h + b2h         4x per-head 64x64
//   gl  = (concat(hl) @ Wc + bc) / clip(T,0.5,5)
//   top2 -> softmax -> scatter
//
// Pre-pass: Wp (f32, [k][n]) -> WpT (bf16, [n][k]) in d_ws (1 MB), so the
// main kernel's B-operand staging needs no conversion and is a clean 2D tile
// copy -> done by the Tensor Data Mover (tensor_load_to_lds + s_wait_tensorcnt,
// 6-arg clang-23 builtin) when available; else register-pipelined b128 copies.
//
// One block = 64 tokens, 256 threads (8 wave32s). All intermediates + bf16
// weight copies stay in LDS (~244 KB). All GEMMs use v_wmma_f32_16x16x32_bf16.
// Phase 1 is double-buffered with one barrier per K-step.
// ---------------------------------------------------------------------------

#define N_TOKENS   16384
#define K_DIM      2048
#define HDIM       256     // N_HEADS * HEAD_DIM
#define N_EXPERTS  64
#define TOK_BLK    64
#define KSTEPS     (K_DIM / 32)

typedef __attribute__((ext_vector_type(16))) __bf16 v16bf;
typedef __attribute__((ext_vector_type(8)))  float  v8f;
typedef __attribute__((ext_vector_type(4)))  unsigned int u32x4;
typedef __attribute__((ext_vector_type(8)))  int    i32x8;
typedef __attribute__((ext_vector_type(4)))  int    i32x4;
typedef __attribute__((ext_vector_type(4)))  float  f32x4;

#if defined(__has_builtin)
#if __has_builtin(__builtin_amdgcn_tensor_load_to_lds) && __has_builtin(__builtin_amdgcn_s_wait_tensorcnt)
#define USE_TDM 1
#endif
#endif
#ifndef USE_TDM
#define USE_TDM 0
#endif

struct Frag32B { u32x4 lo; u32x4 hi; };

// LDS strides (elements). Chosen so every 8-bf16 fragment gather is 16B
// aligned (ds_load_b128 / ds_store_b128) and weight writes avoid conflicts.
#define XS_STRIDE   40    // x stage row: 80B
#define WPT_STRIDE  40    // WpT stage row: 80B (= 64B tile row + 16B TDM pad)
#define BUF_STRIDE  272   // token row of h/hid/logits: 544B
#define W1T_STRIDE  72    // per-(head,col) row of W1T/W2T: 144B
#define WCT_STRIDE  264   // per-col row of WcT: 528B
#define GATE_STRIDE 68    // f32 gate logits row

__device__ __forceinline__ unsigned short f2bf(float f) {
  unsigned int u = __float_as_uint(f);
  u += 0x7fffu + ((u >> 16) & 1u);          // round-to-nearest-even
  return (unsigned short)(u >> 16);
}
__device__ __forceinline__ unsigned int pack2bf(float lo, float hi) {
  return ((unsigned int)f2bf(hi) << 16) | (unsigned int)f2bf(lo);
}

// One 16x32 bf16 WMMA fragment from an LDS row pointer (k=0 of the 32-wide
// K chunk for this lane's row/col). ISA 16-bit A-layout:
//   lanes 0-15 : K 0..7  in v0..3, K 16..23 in v4..7
//   lanes 16-31: K 8..15 in v0..3, K 24..31 in v4..7
// B fragments use the mirrored layout with lanes = columns.
__device__ __forceinline__ v16bf load_frag(const unsigned short* rowbase, int lane) {
  const unsigned short* p = rowbase + ((lane & 16) ? 8 : 0);
  Frag32B f;
  f.lo = *reinterpret_cast<const u32x4*>(p);        // K base+0..7
  f.hi = *reinterpret_cast<const u32x4*>(p + 16);   // K base+16..23
  return __builtin_bit_cast(v16bf, f);
}

__device__ __forceinline__ v8f zero8() {
  v8f v = {0.f, 0.f, 0.f, 0.f, 0.f, 0.f, 0.f, 0.f};
  return v;
}

// ---- x chunk: thread -> token t = tid>>2, 8 consecutive k (two b128 loads)
__device__ __forceinline__ void load_x(const float* __restrict__ x,
                                       int tok0, int k0, int tid, f32x4* xr) {
  const int t = tid >> 2, k8 = (tid & 3) << 3;
  const f32x4* xp = reinterpret_cast<const f32x4*>(x + (size_t)(tok0 + t) * K_DIM + k0 + k8);
  xr[0] = xp[0];
  xr[1] = xp[1];
}
__device__ __forceinline__ void store_x(unsigned short* __restrict__ xs,
                                        int tid, const f32x4* xr) {
  const int t = tid >> 2, k8 = (tid & 3) << 3;
  u32x4 xv = { pack2bf(xr[0][0], xr[0][1]), pack2bf(xr[0][2], xr[0][3]),
               pack2bf(xr[1][0], xr[1][1]), pack2bf(xr[1][2], xr[1][3]) };
  *reinterpret_cast<u32x4*>(&xs[t * XS_STRIDE + k8]) = xv;
}

#if USE_TDM
// Tensor Data Mover: copy the [256 rows x 64B] WpT chunk (global row stride
// 4096B) into LDS rows of 80B (pad_interval = 16 DWORDs, pad_amount = 4 DWORDs).
__device__ __forceinline__ void tdm_load_wp(const unsigned short* __restrict__ WpT,
                                            int k0, unsigned short* ldsdst) {
  unsigned long long ga = (unsigned long long)(uintptr_t)(WpT + k0);
  unsigned int lds = (unsigned int)(uintptr_t)ldsdst;
  u32x4 g0;
  g0[0] = 1u;                                   // count=1 (valid user D#)
  g0[1] = lds;                                  // lds_addr
  g0[2] = (unsigned int)ga;                     // global_addr[31:0]
  g0[3] = (unsigned int)((ga >> 32) & 0x1FFFFFFull) | (2u << 30);  // addr[56:32], type=2
  i32x8 g1;
  g1[0] = (1 << 16)      // data_size = 1 -> 2 bytes
        | (1 << 20)      // pad_enable
        | (3 << 22)      // pad_interval: 16 DWORDs (64B) between pads
        | (3 << 25);     // pad_amount: 4 DWORDs (16B)
  g1[1] = (int)(2048u << 16);                   // tensor_dim0 = 2048 (lo16 in [31:16])
  g1[2] = (int)(256u << 16);                    // dim0 hi = 0 | tensor_dim1 = 256
  g1[3] = (int)(32u << 16);                     // dim1 hi = 0 | tile_dim0 = 32
  g1[4] = 256;                                  // tile_dim1 = 256, tile_dim2 = 0
  g1[5] = 2048;                                 // tensor_dim0_stride = 2048 (lo32)
  g1[6] = 0;
  g1[7] = 0;                                    // tensor_dim1_stride = 0 (2D)
  i32x4 z4 = {0, 0, 0, 0};                      // D# groups 2/3 unused (2D tensor)
  i32x8 z8 = {0, 0, 0, 0, 0, 0, 0, 0};
  __builtin_amdgcn_tensor_load_to_lds(g0, g1, z4, z4, z8, 0);
}
#else
// Fallback: register-pipelined copy of the thread's WpT column (32 bf16 = 64B).
__device__ __forceinline__ void load_wp(const unsigned short* __restrict__ WpT,
                                        int k0, int tid, u32x4* wr) {
  const u32x4* wp = reinterpret_cast<const u32x4*>(WpT + (size_t)tid * K_DIM + k0);
#pragma unroll
  for (int g = 0; g < 4; ++g) wr[g] = wp[g];
}
__device__ __forceinline__ void store_wp(unsigned short* __restrict__ wpT,
                                         int tid, const u32x4* wr) {
#pragma unroll
  for (int g = 0; g < 4; ++g)
    *reinterpret_cast<u32x4*>(&wpT[tid * WPT_STRIDE + 8 * g]) = wr[g];
}
#endif

// ---------------------------------------------------------------------------
// Pre-pass: Wp [2048][256] f32 -> WpT [256][2048] bf16 (LDS-tiled transpose,
// coalesced on both sides). 512 blocks x 256 threads, one 32x32 tile each.
// ---------------------------------------------------------------------------
__global__ __launch_bounds__(256) void wp_transpose_kernel(
    const float* __restrict__ Wp, unsigned short* __restrict__ WpT) {
  __shared__ float tile[32][33];
  const int tid = threadIdx.x;
  const int kt = blockIdx.x & 63;      // 64 tiles along K
  const int nt = blockIdx.x >> 6;      // 8 tiles along N
#pragma unroll
  for (int i = 0; i < 4; ++i) {
    int e = tid + i * 256, r = e >> 5, c = e & 31;
    tile[r][c] = Wp[(size_t)(kt * 32 + r) * HDIM + nt * 32 + c];
  }
  __syncthreads();
#pragma unroll
  for (int i = 0; i < 4; ++i) {
    int e = tid + i * 256, n = e >> 5, k = e & 31;
    WpT[(size_t)(nt * 32 + n) * K_DIM + kt * 32 + k] = f2bf(tile[k][n]);
  }
}

__global__ __launch_bounds__(256) void dyn_gate_kernel(
    const float* __restrict__ x,    const unsigned short* __restrict__ WpT,
    const float* __restrict__ bp,   const float* __restrict__ W1,
    const float* __restrict__ b1,   const float* __restrict__ W2,
    const float* __restrict__ b2,   const float* __restrict__ Wc,
    const float* __restrict__ bc,   const float* __restrict__ temperature,
    float* __restrict__ gates_out,  // [N, 64]
    float* __restrict__ idx_out,    // [N, 2]  (int indices stored as floats)
    float* __restrict__ logits_out) // [N, 64]
{
  __shared__ __align__(16) unsigned short s_xs [2][TOK_BLK * XS_STRIDE];  // 2x  5120B
  __shared__ __align__(16) unsigned short s_wpT[2][HDIM * WPT_STRIDE];    // 2x 20480B
  __shared__ __align__(16) unsigned short s_bufA[TOK_BLK * BUF_STRIDE];   // 34816B  h / head_logits
  __shared__ __align__(16) unsigned short s_bufB[TOK_BLK * BUF_STRIDE];   // 34816B  hid
  __shared__ __align__(16) unsigned short s_w1T[HDIM * W1T_STRIDE];       // 36864B  [h*64+e][d]
  __shared__ __align__(16) unsigned short s_w2T[HDIM * W1T_STRIDE];       // 36864B
  __shared__ __align__(16) unsigned short s_wcT[N_EXPERTS * WCT_STRIDE];  // 33792B  [e][k]
  __shared__ __align__(16) float s_gate[TOK_BLK * GATE_STRIDE];           // 17408B
  __shared__ float s_bp[HDIM], s_b1[HDIM], s_b2[HDIM], s_bc[N_EXPERTS];
  __shared__ float s_g0[TOK_BLK], s_g1[TOK_BLK];
  __shared__ int   s_i0[TOK_BLK], s_i1[TOK_BLK];

  const int tid  = threadIdx.x;
  const int lane = tid & 31;
  const int wv   = tid >> 5;           // 0..7
  const int tok0 = blockIdx.x * TOK_BLK;

  // ---- preload bf16 weights (transposed for column-contiguous B frags) ----
  for (int e = tid; e < HDIM * 64; e += 256) {       // W1[h][d][c] -> w1T[(h*64+c)][d]
    int h = e >> 12, d = (e >> 6) & 63, c = e & 63;
    s_w1T[((h << 6) + c) * W1T_STRIDE + d] = f2bf(W1[e]);
  }
  for (int e = tid; e < HDIM * 64; e += 256) {       // W2 same shape
    int h = e >> 12, d = (e >> 6) & 63, c = e & 63;
    s_w2T[((h << 6) + c) * W1T_STRIDE + d] = f2bf(W2[e]);
  }
  for (int e = tid; e < HDIM * N_EXPERTS; e += 256) { // Wc[k][e] -> wcT[e][k]
    int k = e >> 6, c = e & 63;
    s_wcT[c * WCT_STRIDE + k] = f2bf(Wc[e]);
  }
  if (tid < HDIM) { s_bp[tid] = bp[tid]; s_b1[tid] = b1[tid]; s_b2[tid] = b2[tid]; }
  if (tid < N_EXPERTS) s_bc[tid] = bc[tid];

  // =================== Phase 1: h = x @ Wp + bp (K = 2048) ===================
  // wave wv: row-stripe r = wv&3 (16 tokens), column half chalf = wv>>2.
  // Double-buffered staging; x pipelined through registers, Wp via TDM.
  const int r1    = wv & 3;
  const int chalf = wv >> 2;
  v8f acc[8];
#pragma unroll
  for (int i = 0; i < 8; ++i) acc[i] = zero8();

  f32x4 xr[2];
#if !USE_TDM
  u32x4 wr[4];
#endif

  // prologue: chunk 0
#if USE_TDM
  if (wv == 0) tdm_load_wp(WpT, 0, s_wpT[0]);
#else
  load_wp(WpT, 0, tid, wr);
#endif
  load_x(x, tok0, 0, tid, xr);
  store_x(s_xs[0], tid, xr);
#if USE_TDM
  if (wv == 0) __builtin_amdgcn_s_wait_tensorcnt((short)0);
#else
  store_wp(s_wpT[0], tid, wr);
#endif
  __syncthreads();

  for (int kk = 0; kk < KSTEPS; ++kk) {
    const int cur = kk & 1, nxt = cur ^ 1;
    const bool have_next = (kk + 1 < KSTEPS);
    if (have_next) {
#if USE_TDM
      if (wv == 0) tdm_load_wp(WpT, (kk + 1) << 5, s_wpT[nxt]);   // async DMA
#else
      load_wp(WpT, (kk + 1) << 5, tid, wr);                        // in flight
#endif
      load_x(x, tok0, (kk + 1) << 5, tid, xr);                     // in flight
    }

    v16bf a = load_frag(&s_xs[cur][(r1 * 16 + (lane & 15)) * XS_STRIDE], lane);
    v16bf bfr[8];
#pragma unroll
    for (int nt = 0; nt < 8; ++nt) {
      int col = ((chalf << 3) + nt) * 16 + (lane & 15);
      bfr[nt] = load_frag(&s_wpT[cur][col * WPT_STRIDE], lane);
    }
#pragma unroll
    for (int nt = 0; nt < 8; ++nt)
      acc[nt] = __builtin_amdgcn_wmma_f32_16x16x32_bf16(
          false, a, false, bfr[nt], (short)0, acc[nt], false, false);

    if (have_next) {
      store_x(s_xs[nxt], tid, xr);                                 // drains x loads
#if !USE_TDM
      store_wp(s_wpT[nxt], tid, wr);
#endif
    }
#if USE_TDM
    if (wv == 0) __builtin_amdgcn_s_wait_tensorcnt((short)0);      // drain TDM
#endif
    __syncthreads();
  }

  // bias + store h (bf16) to bufA. C layout: vgpr v -> row v (lanes<16) / v+8.
#pragma unroll
  for (int nt = 0; nt < 8; ++nt) {
    int col = ((chalf << 3) + nt) * 16 + (lane & 15);
    float bv = s_bp[col];
#pragma unroll
    for (int v = 0; v < 8; ++v) {
      int tr = r1 * 16 + v + ((lane & 16) ? 8 : 0);
      s_bufA[tr * BUF_STRIDE + col] = f2bf(acc[nt][v] + bv);
    }
  }
  __syncthreads();

  // ============ Phase 2: hid = relu(h_head @ W1h + b1h)  (64 tile jobs) =====
  for (int i = 0; i < 8; ++i) {
    int job = wv + 8 * i;                       // uniform across waves
    int head = job >> 4, r = (job >> 2) & 3, ct = job & 3;
    v8f c = zero8();
#pragma unroll
    for (int ks = 0; ks < 2; ++ks) {
      v16bf a = load_frag(&s_bufA[(r * 16 + (lane & 15)) * BUF_STRIDE + head * 64 + ks * 32], lane);
      v16bf b = load_frag(&s_w1T[((head << 6) + ct * 16 + (lane & 15)) * W1T_STRIDE + ks * 32], lane);
      c = __builtin_amdgcn_wmma_f32_16x16x32_bf16(false, a, false, b, (short)0, c, false, false);
    }
    int cl = ct * 16 + (lane & 15);
    float bv = s_b1[(head << 6) + cl];
#pragma unroll
    for (int v = 0; v < 8; ++v) {
      int tr = r * 16 + v + ((lane & 16) ? 8 : 0);
      s_bufB[tr * BUF_STRIDE + (head << 6) + cl] = f2bf(fmaxf(c[v] + bv, 0.f));
    }
  }
  __syncthreads();

  // ============ Phase 3: head_logits = hid_head @ W2h + b2h =================
  for (int i = 0; i < 8; ++i) {
    int job = wv + 8 * i;
    int head = job >> 4, r = (job >> 2) & 3, ct = job & 3;
    v8f c = zero8();
#pragma unroll
    for (int ks = 0; ks < 2; ++ks) {
      v16bf a = load_frag(&s_bufB[(r * 16 + (lane & 15)) * BUF_STRIDE + head * 64 + ks * 32], lane);
      v16bf b = load_frag(&s_w2T[((head << 6) + ct * 16 + (lane & 15)) * W1T_STRIDE + ks * 32], lane);
      c = __builtin_amdgcn_wmma_f32_16x16x32_bf16(false, a, false, b, (short)0, c, false, false);
    }
    int cl = ct * 16 + (lane & 15);
    float bv = s_b2[(head << 6) + cl];
#pragma unroll
    for (int v = 0; v < 8; ++v) {
      int tr = r * 16 + v + ((lane & 16) ? 8 : 0);
      s_bufA[tr * BUF_STRIDE + (head << 6) + cl] = f2bf(c[v] + bv);   // overwrite h
    }
  }
  __syncthreads();

  // ====== Phase 4: gate = (combined @ Wc + bc) * 1/clip(T) (K = 256) ========
  float tt = temperature[0];
  tt = fminf(fmaxf(tt, 0.5f), 5.0f);
  const float rtemp = 1.0f / tt;

  for (int i = 0; i < 2; ++i) {
    int job = wv + 8 * i;                       // 16 jobs: 4 rstripes x 4 ctiles
    int r = job >> 2, ct = job & 3;
    v8f c = zero8();
    v16bf bfr2[8];
#pragma unroll
    for (int ks = 0; ks < 8; ++ks)
      bfr2[ks] = load_frag(&s_wcT[(ct * 16 + (lane & 15)) * WCT_STRIDE + ks * 32], lane);
#pragma unroll
    for (int ks = 0; ks < 8; ++ks) {
      v16bf a = load_frag(&s_bufA[(r * 16 + (lane & 15)) * BUF_STRIDE + ks * 32], lane);
      c = __builtin_amdgcn_wmma_f32_16x16x32_bf16(false, a, false, bfr2[ks], (short)0, c, false, false);
    }
    int cl = ct * 16 + (lane & 15);
    float bv = s_bc[cl];
#pragma unroll
    for (int v = 0; v < 8; ++v) {
      int tr = r * 16 + v + ((lane & 16) ? 8 : 0);
      s_gate[tr * GATE_STRIDE + cl] = (c[v] + bv) * rtemp;
    }
  }
  __syncthreads();

  // ============ Phase 5a: per-token top-2 + 2-way softmax ==================
  if (tid < TOK_BLK) {
    const float* g = &s_gate[tid * GATE_STRIDE];
    float v0 = -3.4e38f; int i0 = 0;
    for (int e = 0; e < N_EXPERTS; ++e) { float v = g[e]; if (v > v0) { v0 = v; i0 = e; } }
    float v1 = -3.4e38f; int i1 = 0;
    for (int e = 0; e < N_EXPERTS; ++e) { if (e == i0) continue; float v = g[e]; if (v > v1) { v1 = v; i1 = e; } }
    float e1 = __expf(v1 - v0);          // v0 >= v1
    float inv = 1.0f / (1.0f + e1);
    s_i0[tid] = i0; s_i1[tid] = i1;
    s_g0[tid] = inv; s_g1[tid] = e1 * inv;
    idx_out[(size_t)(tok0 + tid) * 2 + 0] = (float)i0;
    idx_out[(size_t)(tok0 + tid) * 2 + 1] = (float)i1;
  }
  __syncthreads();

  // ===== Phase 5b: coalesced f32x4 stores of gates and gate_logits =========
  {
    const int t = tid >> 2, q = tid & 3;
    const int i0 = s_i0[t], i1 = s_i1[t];
    const float g0 = s_g0[t], g1 = s_g1[t];
    float* go = &gates_out [(size_t)(tok0 + t) * N_EXPERTS + q * 16];
    float* lo = &logits_out[(size_t)(tok0 + t) * N_EXPERTS + q * 16];
    const float* gl = &s_gate[t * GATE_STRIDE + q * 16];
#pragma unroll
    for (int j = 0; j < 4; ++j) {
      f32x4 lv = *reinterpret_cast<const f32x4*>(gl + j * 4);
      f32x4 gv;
#pragma unroll
      for (int u = 0; u < 4; ++u) {
        int e = q * 16 + j * 4 + u;
        gv[u] = (e == i0) ? g0 : ((e == i1) ? g1 : 0.f);
      }
      *reinterpret_cast<f32x4*>(go + j * 4) = gv;
      *reinterpret_cast<f32x4*>(lo + j * 4) = lv;
    }
  }
}

extern "C" void kernel_launch(void* const* d_in, const int* in_sizes, int n_in,
                              void* d_out, int out_size, void* d_ws, size_t ws_size,
                              hipStream_t stream) {
  (void)in_sizes; (void)n_in; (void)ws_size; (void)out_size;
  const float* x   = (const float*)d_in[0];
  const float* Wp  = (const float*)d_in[1];
  const float* bp  = (const float*)d_in[2];
  const float* W1  = (const float*)d_in[3];
  const float* b1  = (const float*)d_in[4];
  const float* W2  = (const float*)d_in[5];
  const float* b2  = (const float*)d_in[6];
  const float* Wc  = (const float*)d_in[7];
  const float* bc  = (const float*)d_in[8];
  const float* temp= (const float*)d_in[9];
  // d_in[10] = noise_scale (unused in eval-mode forward)

  unsigned short* WpT = (unsigned short*)d_ws;              // [256][2048] bf16, 1 MB

  float* out       = (float*)d_out;
  float* gates     = out;                                   // [N, 64]
  float* idx       = out + (size_t)N_TOKENS * N_EXPERTS;    // [N, 2] as float
  float* logits    = idx + (size_t)N_TOKENS * 2;            // [N, 64]

  wp_transpose_kernel<<<dim3(512), dim3(256), 0, stream>>>(Wp, WpT);

  dim3 grid(N_TOKENS / TOK_BLK);   // 256 blocks
  dim3 block(256);                 // 8 wave32s
  dyn_gate_kernel<<<grid, block, 0, stream>>>(x, WpT, bp, W1, b1, W2, b2, Wc, bc,
                                              temp, gates, idx, logits);
}
